// _MultiQueryAttentionBase_51634096833172
// MI455X (gfx1250) — matrix operand order repaired and checked by reading, not compiled, not run on hardware
//
#include <hip/hip_runtime.h>

// ---------------------------------------------------------------------------
// MI455X (gfx1250) multi-query attention, bf16 WMMA pipeline.
// Stages: f32->bf16 convert | QKV GEMM (WMMA, async-LDS staging) | RoPE |
//         flash-attn (WMMA, TDM K staging, transposed V in LDS) |
//         out GEMM (WMMA, f32 store).
// ---------------------------------------------------------------------------

typedef unsigned short u16;
typedef __attribute__((ext_vector_type(16))) __bf16        v16bf;
typedef __attribute__((ext_vector_type(16))) unsigned short v16u;
typedef __attribute__((ext_vector_type(8)))  float          v8f;
typedef __attribute__((ext_vector_type(4)))  unsigned       v4u;
typedef __attribute__((ext_vector_type(8)))  unsigned       v8u;

#define HH     4          // query heads
#define DD     256        // head dim
#define HSIZE  640        // hidden size
#define BB     4          // batch
#define TT     4096       // seq len
#define NTOK   (BB * TT)  // 16384
#define QDIM   (HH * DD)  // 1024
#define WIN    512
#define QSCALE 0.0625f    // D^-0.5

__device__ __forceinline__ u16 f2bf(float f) {
  unsigned int u = __builtin_bit_cast(unsigned int, f);
  u += 0x7FFFu + ((u >> 16) & 1u);          // round-to-nearest-even
  return (u16)(u >> 16);
}
__device__ __forceinline__ float bf2f(u16 h) {
  unsigned int u = ((unsigned int)h) << 16;
  return __builtin_bit_cast(float, u);
}
// A-matrix (16x32, 16-bit) element->K mapping per ISA 7.12.2
__device__ __forceinline__ int akidx(int i, int half) {
  return (i < 8) ? (i + half * 8) : (i + 8 + half * 8);
}

// CDNA5 async global->LDS copy (16B per lane), tracked by ASYNCcnt.
__device__ __forceinline__ void async_load_b128(unsigned lds_off,
                                                const u16* gaddr) {
  asm volatile("global_load_async_to_lds_b128 %0, %1, off"
               :: "v"(lds_off), "v"((unsigned long long)(size_t)gaddr)
               : "memory");
}
__device__ __forceinline__ void wait_async0() {
  asm volatile("s_wait_asynccnt 0x0" ::: "memory");
}

// CDNA5 Tensor Data Mover: DMA a 2D bf16 tile [32 rows x 256 cols],
// row stride 256 elements, global -> LDS.  D# per ISA 08_async_tensor §8.
__device__ __forceinline__ void tdm_load_k_tile(unsigned lds_byte_addr,
                                                const u16* gsrc) {
  unsigned long long ga = (unsigned long long)(size_t)gsrc;
  v4u g0;
  g0[0] = 1u;                                      // count=1, no gather
  g0[1] = lds_byte_addr;                           // lds_addr
  g0[2] = (unsigned)ga;                            // global_addr[31:0]
  g0[3] = (unsigned)((ga >> 32) & 0x01FFFFFFu)     // global_addr[56:32]
          | 0x80000000u;                           // type=2 ("image")
  v8u g1;
  g1[0] = 0x00010000u;       // workgroup_mask=0, data_size=1 (2 bytes)
  g1[1] = 256u << 16;        // tensor_dim0 = 256 (bits 79:48, low half)
  g1[2] = 32u << 16;         // tensor_dim1 = 32  (bits 111:80, low half)
  g1[3] = 256u << 16;        // tile_dim0   = 256 (bits 127:112)
  g1[4] = 32u;               // tile_dim1   = 32  (bits 143:128); tile_dim2=0
  g1[5] = 256u;              // tensor_dim0_stride = 256 (bits 207:160)
  g1[6] = 0u;
  g1[7] = 0u;                // tensor_dim1_stride = 0 (2D)
  asm volatile("tensor_load_to_lds %0, %1" :: "s"(g0), "s"(g1) : "memory");
}

// ---------------------------------------------------------------------------
// fp32 -> bf16 convert
// ---------------------------------------------------------------------------
__global__ __launch_bounds__(256) void cvt_f32_bf16(const float* __restrict__ in,
                                                    u16* __restrict__ out, int n) {
  int i = blockIdx.x * 256 + threadIdx.x;
  if (i < n) out[i] = f2bf(in[i]);
}

// ---------------------------------------------------------------------------
// RoPE in place on bf16 [tok, heads, 256]; outscale folds q-scaling.
// ---------------------------------------------------------------------------
__global__ __launch_bounds__(256) void rope_bf16(u16* __restrict__ X, int heads,
                                                 float outscale, int total) {
  int idx = blockIdx.x * 256 + threadIdx.x;
  if (idx >= total) return;
  int d    = idx & 127;
  int rest = idx >> 7;
  int hh   = rest % heads;
  int tok  = rest / heads;
  int pos  = tok & (TT - 1);
  u16* p = X + ((size_t)tok * heads + hh) * DD;
  float x1 = bf2f(p[d]);
  float x2 = bf2f(p[d + 128]);
  float fi = (float)(d >> 1);
  const float c = 9.210340371976184f / 128.0f;   // ln(10000)/128
  float a1 = (float)pos * __expf(-fi * c);
  float a2 = (float)pos * __expf(-(fi + 64.0f) * c);
  float o1 = x1 * cosf(a1) - x2 * sinf(a1);
  float o2 = x2 * cosf(a2) + x1 * sinf(a2);
  p[d]       = f2bf(o1 * outscale);
  p[d + 128] = f2bf(o2 * outscale);
}

// ---------------------------------------------------------------------------
// Y[M,N] = A[M,Kd] @ W[N,Kd]^T, bf16 in, f32 accum, bf16 or f32 out.
// Block tile 128x64, 8 waves each 32x32 (2x2 frags), BK=64 (8 WMMA/stage).
// Tiles staged with async global->LDS copies (unconditional unrolled issues).
// ---------------------------------------------------------------------------
__global__ __launch_bounds__(256) void gemm_bf16_wmma(
    const u16* __restrict__ A, const u16* __restrict__ W,
    u16* __restrict__ Ybf, float* __restrict__ Yf,
    int M, int N, int Kd, int store_f32) {
  __shared__ u16 As[128 * 64];   // 16 KB
  __shared__ u16 Ws[64 * 64];    // 8 KB
  const int tid  = threadIdx.x;
  const int lane = tid & 31;
  const int wave = tid >> 5;
  const int half = lane >> 4;
  const int l16  = lane & 15;
  const int bm0  = blockIdx.x * 128;
  const int bn0  = blockIdx.y * 64;
  const int wm0  = (wave >> 1) * 32;   // 4 wave-rows
  const int wn0  = (wave & 1) * 32;    // 2 wave-cols
  const unsigned asBase = (unsigned)(size_t)As;
  const unsigned wsBase = (unsigned)(size_t)Ws;

  v8f acc[2][2];
#pragma unroll
  for (int a = 0; a < 2; ++a)
#pragma unroll
    for (int b = 0; b < 2; ++b)
#pragma unroll
      for (int r = 0; r < 8; ++r) acc[a][b][r] = 0.f;

  for (int k0 = 0; k0 < Kd; k0 += 64) {
    // A tile (128x64): 1024 16B chunks, 4 per thread, async to LDS
#pragma unroll
    for (int it = 0; it < 4; ++it) {
      int c = tid + it * 256;
      int row = c >> 3, kc = c & 7;
      async_load_b128(asBase + (unsigned)(row * 64 + kc * 8) * 2u,
                      &A[(size_t)(bm0 + row) * Kd + k0 + kc * 8]);
    }
    // W tile (64x64): 512 chunks, 2 per thread
#pragma unroll
    for (int it = 0; it < 2; ++it) {
      int c = tid + it * 256;
      int row = c >> 3, kc = c & 7;
      async_load_b128(wsBase + (unsigned)(row * 64 + kc * 8) * 2u,
                      &W[(size_t)(bn0 + row) * Kd + k0 + kc * 8]);
    }
    wait_async0();
    __syncthreads();

#pragma unroll
    for (int ks2 = 0; ks2 < 2; ++ks2) {
      v16bf a[2], b[2];
#pragma unroll
      for (int im = 0; im < 2; ++im) {
        v16u t;
#pragma unroll
        for (int i = 0; i < 16; ++i)
          t[i] = As[(wm0 + im * 16 + l16) * 64 + ks2 * 32 + akidx(i, half)];
        a[im] = __builtin_bit_cast(v16bf, t);
      }
#pragma unroll
      for (int in = 0; in < 2; ++in) {
        v16u t;
#pragma unroll
        for (int i = 0; i < 16; ++i)
          t[i] = Ws[(wn0 + in * 16 + l16) * 64 + ks2 * 32 + akidx(i, half)];
        b[in] = __builtin_bit_cast(v16bf, t);
      }
#pragma unroll
      for (int im = 0; im < 2; ++im)
#pragma unroll
        for (int in = 0; in < 2; ++in)
          acc[im][in] = __builtin_amdgcn_wmma_f32_16x16x32_bf16(
              false, a[im], false, b[in], (short)0, acc[im][in], false, false);
    }
    __syncthreads();
  }

#pragma unroll
  for (int im = 0; im < 2; ++im)
#pragma unroll
    for (int in = 0; in < 2; ++in)
#pragma unroll
      for (int r = 0; r < 8; ++r) {
        int row = bm0 + wm0 + im * 16 + r + 8 * half;  // C layout
        int col = bn0 + wn0 + in * 16 + l16;
        float val = acc[im][in][r];
        if (store_f32) Yf[(size_t)row * N + col] = val;
        else           Ybf[(size_t)row * N + col] = f2bf(val);
      }
}

// ---------------------------------------------------------------------------
// Flash attention: block = 4 waves, 64-query tile for one (b,h).
// K tile DMA'd by the Tensor Data Mover (wave 0 issues, TENSORcnt wait);
// V staged TRANSPOSED (Vt[d][key]) so PV B-fragments are contiguous.
// ---------------------------------------------------------------------------
__global__ __launch_bounds__(128) void flash_attn(
    const u16* __restrict__ Q, const u16* __restrict__ Km,
    const u16* __restrict__ Vm, u16* __restrict__ Ctx) {
  __shared__ u16 Ks[32 * 256];   // 16 KB, [key][d]
  __shared__ u16 Vt[256 * 32];   // 16 KB, [d][key]  (transposed V tile)
  __shared__ u16 Ps[4][16 * 32]; // per-wave P scratch (C->A layout bounce)
  const int tid  = threadIdx.x;
  const int wave = tid >> 5;
  const int lane = tid & 31;
  const int half = lane >> 4;
  const int l16  = lane & 15;
  const int q0   = blockIdx.x * 64;
  const int bh   = blockIdx.y;
  const int b    = bh >> 2, h = bh & 3;
  const int bT   = b * TT;
  const int qw   = q0 + wave * 16;
  const unsigned ksBase = (unsigned)(size_t)Ks;

  // Q fragments: 16 queries x 256 dims = 8 A-frags (K=32 each), loaded once.
  v16bf qf[8];
#pragma unroll
  for (int ks = 0; ks < 8; ++ks) {
    v16u t;
#pragma unroll
    for (int i = 0; i < 16; ++i)
      t[i] = Q[(size_t)(bT + qw + l16) * QDIM + h * DD + ks * 32 + akidx(i, half)];
    qf[ks] = __builtin_bit_cast(v16bf, t);
  }

  v8f acc[16];
#pragma unroll
  for (int t2 = 0; t2 < 16; ++t2)
#pragma unroll
    for (int r = 0; r < 8; ++r) acc[t2][r] = 0.f;
  float rmax[8], rsum[8];
#pragma unroll
  for (int r = 0; r < 8; ++r) { rmax[r] = -3.0e38f; rsum[r] = 0.f; }

  int kst = q0 - WIN; if (kst < 0) kst = 0;
  const int kt0 = kst >> 5;
  const int kt1 = (q0 + 63) >> 5;

  for (int kt = kt0; kt <= kt1; ++kt) {
    const int ktb = kt << 5;
    // K tile: one TDM descriptor moves the whole 32x256 bf16 tile.
    if (wave == 0) {
      tdm_load_k_tile(ksBase, &Km[(size_t)(bT + ktb) * DD]);
    }
    // V tile: register transpose -> Vt[d][key], 8 chunks per thread.
#pragma unroll
    for (int it = 0; it < 8; ++it) {
      int c = tid + it * 128;
      int row = c >> 5, dc = (c & 31) << 3;
      uint4 pkt = *(const uint4*)&Vm[(size_t)(bT + ktb + row) * DD + dc];
      const u16* pv = (const u16*)&pkt;
#pragma unroll
      for (int j2 = 0; j2 < 8; ++j2)
        Vt[(dc + j2) * 32 + row] = pv[j2];
    }
    if (kt < kt1) {  // prefetch next K/V tile rows (global_prefetch_b8)
      int row = tid & 31;
      __builtin_prefetch(&Km[(size_t)(bT + ktb + 32 + row) * DD], 0, 1);
      __builtin_prefetch(&Vm[(size_t)(bT + ktb + 32 + row) * DD], 0, 1);
    }
    if (wave == 0) __builtin_amdgcn_s_wait_tensorcnt(0);
    __syncthreads();

    // S = Q @ K^T for 2 key sub-tiles of 16
    v8f s[2];
#pragma unroll
    for (int j = 0; j < 2; ++j)
#pragma unroll
      for (int r = 0; r < 8; ++r) s[j][r] = 0.f;
#pragma unroll
    for (int j = 0; j < 2; ++j)
#pragma unroll
      for (int ks = 0; ks < 8; ++ks) {
        v16u t;
#pragma unroll
        for (int i = 0; i < 16; ++i)
          t[i] = Ks[(j * 16 + l16) * 256 + ks * 32 + akidx(i, half)];
        v16bf bfr = __builtin_bit_cast(v16bf, t);
        s[j] = __builtin_amdgcn_wmma_f32_16x16x32_bf16(
            false, qf[ks], false, bfr, (short)0, s[j], false, false);
      }

    // causal + sliding-window mask
#pragma unroll
    for (int j = 0; j < 2; ++j)
#pragma unroll
      for (int r = 0; r < 8; ++r) {
        int key = ktb + j * 16 + l16;
        int qq  = qw + r + 8 * half;
        bool ok = (key <= qq) && (qq - key <= WIN);
        if (!ok) s[j][r] = -3.0e38f;
      }

    // online softmax; row = (r, half); reduce across the 16 lanes of half
    float sc[8];
#pragma unroll
    for (int r = 0; r < 8; ++r) {
      float lm = fmaxf(s[0][r], s[1][r]);
#pragma unroll
      for (int m = 1; m < 16; m <<= 1) lm = fmaxf(lm, __shfl_xor(lm, m, 32));
      float nm = fmaxf(rmax[r], lm);
      float scale = (nm > -1.0e37f) ? __expf(rmax[r] - nm) : 1.0f;
      float ps = 0.f;
#pragma unroll
      for (int j = 0; j < 2; ++j) {
        float sv = s[j][r];
        float p  = (sv > -1.0e37f) ? __expf(sv - nm) : 0.f;
        s[j][r] = p;
        ps += p;
      }
#pragma unroll
      for (int m = 1; m < 16; m <<= 1) ps += __shfl_xor(ps, m, 32);
      rsum[r] = rsum[r] * scale + ps;
      rmax[r] = nm;
      sc[r]   = scale;
    }
#pragma unroll
    for (int t2 = 0; t2 < 16; ++t2)
#pragma unroll
      for (int r = 0; r < 8; ++r) acc[t2][r] *= sc[r];

    // P: C-layout -> bf16 -> per-wave LDS -> A-layout frag (16q x 32keys)
#pragma unroll
    for (int j = 0; j < 2; ++j)
#pragma unroll
      for (int r = 0; r < 8; ++r)
        Ps[wave][(r + 8 * half) * 32 + j * 16 + l16] = f2bf(s[j][r]);
    v16bf pf;
    {
      v16u t;
#pragma unroll
      for (int i = 0; i < 16; ++i) t[i] = Ps[wave][l16 * 32 + akidx(i, half)];
      pf = __builtin_bit_cast(v16bf, t);
    }

    // acc += P @ V across 16 dim sub-tiles; B-frag contiguous from Vt
#pragma unroll
    for (int t2 = 0; t2 < 16; ++t2) {
      v16u t;
#pragma unroll
      for (int i = 0; i < 16; ++i)
        t[i] = Vt[(t2 * 16 + l16) * 32 + akidx(i, half)];
      v16bf bfr = __builtin_bit_cast(v16bf, t);
      acc[t2] = __builtin_amdgcn_wmma_f32_16x16x32_bf16(
          false, pf, false, bfr, (short)0, acc[t2], false, false);
    }
    __syncthreads();
  }

  float inv[8];
#pragma unroll
  for (int r = 0; r < 8; ++r) inv[r] = 1.0f / rsum[r];
#pragma unroll
  for (int t2 = 0; t2 < 16; ++t2)
#pragma unroll
    for (int r = 0; r < 8; ++r) {
      int qq = qw + r + 8 * half;
      int d  = t2 * 16 + l16;
      Ctx[(size_t)(bT + qq) * QDIM + h * DD + d] = f2bf(acc[t2][r] * inv[r]);
    }
}

// ---------------------------------------------------------------------------
extern "C" void kernel_launch(void* const* d_in, const int* in_sizes, int n_in,
                              void* d_out, int out_size, void* d_ws, size_t ws_size,
                              hipStream_t stream) {
  (void)in_sizes; (void)n_in; (void)out_size; (void)ws_size;
  const float* x  = (const float*)d_in[0];
  const float* wq = (const float*)d_in[1];
  const float* wk = (const float*)d_in[2];
  const float* wv = (const float*)d_in[3];
  const float* wo = (const float*)d_in[4];
  float* out = (float*)d_out;

  char* ws = (char*)d_ws;
  size_t off = 0;
  auto take = [&](size_t elems) -> u16* {
    u16* p = (u16*)(ws + off);
    off += ((elems * 2 + 255) & ~(size_t)255);
    return p;
  };
  u16* x_bf  = take((size_t)NTOK * HSIZE);
  u16* wq_bf = take((size_t)QDIM * HSIZE);
  u16* wk_bf = take((size_t)DD * HSIZE);
  u16* wv_bf = take((size_t)DD * HSIZE);
  u16* wo_bf = take((size_t)HSIZE * QDIM);
  u16* q_bf  = take((size_t)NTOK * QDIM);
  u16* k_bf  = take((size_t)NTOK * DD);
  u16* v_bf  = take((size_t)NTOK * DD);
  u16* c_bf  = take((size_t)NTOK * QDIM);   // total ~108 MB of d_ws

  auto cvt = [&](const float* src, u16* dst, int n) {
    cvt_f32_bf16<<<(n + 255) / 256, 256, 0, stream>>>(src, dst, n);
  };
  cvt(x,  x_bf,  NTOK * HSIZE);
  cvt(wq, wq_bf, QDIM * HSIZE);
  cvt(wk, wk_bf, DD * HSIZE);
  cvt(wv, wv_bf, DD * HSIZE);
  cvt(wo, wo_bf, HSIZE * QDIM);

  gemm_bf16_wmma<<<dim3(NTOK / 128, QDIM / 64), 256, 0, stream>>>(
      x_bf, wq_bf, q_bf, nullptr, NTOK, QDIM, HSIZE, 0);
  gemm_bf16_wmma<<<dim3(NTOK / 128, DD / 64), 256, 0, stream>>>(
      x_bf, wk_bf, k_bf, nullptr, NTOK, DD, HSIZE, 0);
  gemm_bf16_wmma<<<dim3(NTOK / 128, DD / 64), 256, 0, stream>>>(
      x_bf, wv_bf, v_bf, nullptr, NTOK, DD, HSIZE, 0);

  rope_bf16<<<(NTOK * HH * 128 + 255) / 256, 256, 0, stream>>>(
      q_bf, HH, QSCALE, NTOK * HH * 128);
  rope_bf16<<<(NTOK * 128 + 255) / 256, 256, 0, stream>>>(
      k_bf, 1, 1.0f, NTOK * 128);

  flash_attn<<<dim3(TT / 64, BB * HH), 128, 0, stream>>>(q_bf, k_bf, v_bf, c_bf);

  gemm_bf16_wmma<<<dim3(NTOK / 128, HSIZE / 64), 256, 0, stream>>>(
      c_bf, wo_bf, nullptr, out, NTOK, HSIZE, QDIM, 1);
}
